// TragetAttentionPooling_9096740733058
// MI455X (gfx1250) — matrix operand order repaired and checked by reading, not compiled, not run on hardware
//
#include <hip/hip_runtime.h>
#include <hip/hip_bf16.h>
#include <math.h>

#define FDIM 128
#define HDIM 128
#define NWAVES 8
#define TM 16
#define XPAD 4            // 4 dwords of LDS padding per 128-dword row (TDM pad feature)

typedef __attribute__((ext_vector_type(16))) __bf16 bf16x16;
typedef __attribute__((ext_vector_type(8)))  float  v8f;
typedef __attribute__((ext_vector_type(4)))  unsigned int u32x4;
typedef __attribute__((ext_vector_type(8)))  unsigned int u32x8;

// ---------------------------------------------------------------------------
// WMMA helper: D = A(16x32 bf16) * B(32x16 bf16) + C(16x16 f32)
// ---------------------------------------------------------------------------
__device__ __forceinline__ v8f wmma_bf16(bf16x16 a, bf16x16 b, v8f c) {
  return __builtin_amdgcn_wmma_f32_16x16x32_bf16(
      /*neg_a=*/false, a, /*neg_b=*/false, b,
      /*c_mod=*/(short)0, c, /*reuse_a=*/false, /*reuse_b=*/false);
}

// A-fragment from f32 LDS row, converting to bf16 in flight.  Contiguous k order (this code's choice):
// elem j -> k = 32*ks + 16*half + j
__device__ __forceinline__ bf16x16 load_a_f32(const float* row, int ks, int half) {
  bf16x16 a;
  const int kb = (ks << 5) + (half << 4);
#pragma unroll
  for (int j = 0; j < 8; ++j) a[j] = (__bf16)row[kb + j];
#pragma unroll
  for (int j = 0; j < 8; ++j) a[8 + j] = (__bf16)row[kb + 8 + j];
  return a;
}

// B-fragment, pre-packed in workspace so each lane reads 32 contiguous bytes.
__device__ __forceinline__ bf16x16 load_bfrag(const unsigned short* wpack,
                                              int mat, int ks, int t, int lane) {
  const bf16x16* p = (const bf16x16*)wpack;
  return p[mat * 1024 + ((ks << 3) + t) * 32 + lane];
}

// ---------------------------------------------------------------------------
// TDM: DMA one 16x128 f32 tile (rows beyond rem_rows auto-zero via OOB) into
// LDS with 4-dword padding after each 128-dword row.  D# per ISA ch.8.
// Issued by one wave; tracked with TENSORcnt.
// ---------------------------------------------------------------------------
__device__ __forceinline__ void tdm_load_tile(unsigned lds_addr,
                                              unsigned long long gaddr,
                                              unsigned rem_rows) {
  u32x4 g0;
  u32x8 g1;
  g0[0] = 1u;                                                    // count=1 (valid D#)
  g0[1] = lds_addr;                                              // lds_addr[31:0]
  g0[2] = (unsigned)gaddr;                                       // global_addr[31:0]
  g0[3] = ((unsigned)(gaddr >> 32) & 0x01FFFFFFu) | (2u << 30);  // addr[56:32] | type=2
  g1[0] = (2u << 16)        // data_size = 4 bytes
        | (1u << 20)        // pad_enable
        | (6u << 22)        // pad_interval: 128 dwords (one row)
        | (3u << 25);       // pad_amount: 4 dwords
  g1[1] = (unsigned)FDIM << 16;                                  // tensor_dim0 = 128
  g1[2] = (rem_rows & 0xFFFFu) << 16;                            // tensor_dim1[15:0]
  g1[3] = ((unsigned)FDIM << 16) | ((rem_rows >> 16) & 0xFFFFu); // tile_dim0=128 | dim1 hi
  g1[4] = (unsigned)TM;                                          // tile_dim1=16, tile_dim2=0
  g1[5] = (unsigned)FDIM;                                        // tensor_dim0_stride = 128
  g1[6] = 0u;
  g1[7] = 0u;
  asm volatile("tensor_load_to_lds %0, %1" :: "s"(g0), "s"(g1) : "memory");
}

// ---------------------------------------------------------------------------
// Kernel 1: first[b] = searchsorted(segment_ids, b); first[B] = N
// ---------------------------------------------------------------------------
__global__ void build_first_kernel(const int* __restrict__ seg,
                                   int* __restrict__ first, int Bn, int N) {
  int b = blockIdx.x * blockDim.x + threadIdx.x;
  if (b > Bn) return;
  if (b == Bn) { first[b] = N; return; }
  int lo = 0, hi = N;
  while (lo < hi) {
    int mid = (lo + hi) >> 1;
    if (seg[mid] < b) lo = mid + 1; else hi = mid;
  }
  first[b] = lo;
}

// ---------------------------------------------------------------------------
// Kernel 2: pack one 128x128 f32 weight into bf16 WMMA-B fragment layout:
// dst[((ks*8 + t)*32 + lane)*16 + j] = W[32*ks + 16*(lane/16) + j][16*t + lane%16]
// ---------------------------------------------------------------------------
__global__ void pack_weight_kernel(const float* __restrict__ W,
                                   unsigned short* __restrict__ dst) {
  int idx = blockIdx.x * 256 + threadIdx.x;   // 16384 elements
  int j    = idx & 15;
  int lane = (idx >> 4) & 31;
  int t    = (idx >> 9) & 7;
  int ks   = idx >> 12;
  int k = (ks << 5) + ((lane >> 4) << 4) + j;
  int n = (t << 4) + (lane & 15);
  __bf16 v = (__bf16)W[k * HDIM + n];
  dst[idx] = __builtin_bit_cast(unsigned short, v);
}

// ---------------------------------------------------------------------------
// Kernel 3: fused per-segment attention pooling.  One workgroup (8 wave32) per
// graph; wave w owns H-columns [16w,16w+16).  TDM double-buffers node tiles
// into LDS; online softmax over tiles.
// Packed weight order: 0=i_Wq 1=u_Wq 2=i_Wk 3=i_Wv 4=u_Wk 5=u_Wv
// ---------------------------------------------------------------------------
__global__ __launch_bounds__(256)
void seg_attn_kernel(const float* __restrict__ feat,
                     const int* __restrict__ first,
                     const unsigned short* __restrict__ wpack,
                     const float* __restrict__ i_bq, const float* __restrict__ i_bk,
                     const float* __restrict__ i_bv,
                     const float* __restrict__ u_bq, const float* __restrict__ u_bk,
                     const float* __restrict__ u_bv,
                     float* __restrict__ out, int Bn) {
  const int tid  = threadIdx.x;
  const int lane = tid & 31;
  const int w    = tid >> 5;          // wave id 0..7
  const int half = lane >> 4;         // 0: rows 0-7 / K-lo, 1: rows 8-15 / K-hi
  const int l16  = lane & 15;
  const int col  = (w << 4) | l16;    // this lane's H column
  const int b    = blockIdx.x;

  __shared__ float  Xf[2][TM][FDIM + XPAD];  // double-buffered f32 node tiles (TDM dest)
  __shared__ float  qi_s[HDIM], qu_s[HDIM];  // per-segment query vectors
  __shared__ float  red[2][TM][NWAVES];      // per-wave partial scores
  __shared__ float  scr[2][TM];              // full scores for tile
  __shared__ float  esc[2][TM];              // exp(score - running max)
  __shared__ float  mrun[2], srun[2], scl[2];

  const int seg_start = first[b];
  const int seg_len   = first[b + 1] - seg_start;
  const int ntiles    = (seg_len + TM - 1) >> 4;

  const unsigned lds0 = (unsigned)(size_t)&Xf[0][0][0];
  const unsigned lds1 = (unsigned)(size_t)&Xf[1][0][0];
  const unsigned long long gbase =
      (unsigned long long)(size_t)(feat + (size_t)seg_start * FDIM);

  // ---- kick off TDM for tiles 0 (and 1), wait for tile 0 ----
  if (w == 0) {
    tdm_load_tile(lds0, gbase, (unsigned)seg_len);
    if (ntiles > 1) {
      tdm_load_tile(lds1, gbase + (unsigned long long)TM * FDIM * 4,
                    (unsigned)(seg_len - TM));
      __builtin_amdgcn_s_wait_tensorcnt(1);
    } else {
      __builtin_amdgcn_s_wait_tensorcnt(0);
    }
  }

  const float bki = i_bk[col], bku = u_bk[col];
  const float bvi = i_bv[col], bvu = u_bv[col];
  if (tid < 2) { mrun[tid] = -__builtin_inff(); srun[tid] = 0.f; }
  __syncthreads();   // tile 0 visible to all waves

  // ---- q_i = feat[first]@i_Wq + b, q_u = feat[first+1]@u_Wq + b (via WMMA) ----
  {
    const float* arow = &Xf[0][l16][0];
    v8f aqi = {}, aqu = {};
#pragma unroll
    for (int ks = 0; ks < 4; ++ks) {
      bf16x16 a = load_a_f32(arow, ks, half);
      aqi = wmma_bf16(a, load_bfrag(wpack, 0, ks, w, lane), aqi);
      aqu = wmma_bf16(a, load_bfrag(wpack, 1, ks, w, lane), aqu);
    }
    if (half == 0) {                 // D layout: VGPR r, lanes 0-15 -> M=r
      qi_s[col] = aqi[0] + i_bq[col];   // M=0 : item target row
      qu_s[col] = aqu[1] + u_bq[col];   // M=1 : user target row
    }
  }
  __syncthreads();

  const float qiv = qi_s[col];
  const float quv = qu_s[col];

  float acc_oi = 0.f, acc_ou = 0.f;  // online weighted-v accumulators (this col)

  for (int t = 0; t < ntiles; ++t) {
    // ---- 4 fused projections: 16 WMMAs per wave per tile ----
    const float* arow = &Xf[t & 1][l16][0];
    v8f aik = {}, aiv = {}, auk = {}, auv = {};
#pragma unroll
    for (int ks = 0; ks < 4; ++ks) {
      bf16x16 a = load_a_f32(arow, ks, half);
      aik = wmma_bf16(a, load_bfrag(wpack, 2, ks, w, lane), aik);
      aiv = wmma_bf16(a, load_bfrag(wpack, 3, ks, w, lane), aiv);
      auk = wmma_bf16(a, load_bfrag(wpack, 4, ks, w, lane), auk);
      auv = wmma_bf16(a, load_bfrag(wpack, 5, ks, w, lane), auv);
    }

    // ---- partial scores q·k over this wave's 16 columns ----
#pragma unroll
    for (int r = 0; r < 8; ++r) {
      float si = (aik[r] + bki) * qiv;
      float su = (auk[r] + bku) * quv;
#pragma unroll
      for (int m = 1; m < 16; m <<= 1) {
        si += __shfl_xor(si, m, 32);
        su += __shfl_xor(su, m, 32);
      }
      if (l16 == 0) {
        red[0][r + (half << 3)][w] = si;
        red[1][r + (half << 3)][w] = su;
      }
    }
    __syncthreads();

    if (tid < 32) {   // combine waves -> full scores; mask padded rows
      int sel = tid >> 4, nm = tid & 15;
      float s = 0.f;
#pragma unroll
      for (int ww = 0; ww < NWAVES; ++ww) s += red[sel][nm][ww];
      scr[sel][nm] = ((t << 4) + nm < seg_len) ? s : -__builtin_inff();
    }
    __syncthreads();

    if (tid < 2) {    // online softmax bookkeeping (one thread per i/u)
      int sel = tid;
      float mt = -__builtin_inff();
#pragma unroll
      for (int nm = 0; nm < TM; ++nm) mt = fmaxf(mt, scr[sel][nm]);
      float mo = mrun[sel];
      float mn = fmaxf(mo, mt);
      float sc = (mo == -__builtin_inff()) ? 0.f : __expf(mo - mn);
      float s  = srun[sel] * sc;
#pragma unroll
      for (int nm = 0; nm < TM; ++nm) {
        float sv = scr[sel][nm];
        float e  = (sv == -__builtin_inff()) ? 0.f : __expf(sv - mn);
        esc[sel][nm] = e;
        s += e;
      }
      mrun[sel] = mn; srun[sel] = s; scl[sel] = sc;
    }
    __syncthreads();

    // ---- rescale and accumulate weighted v ----
    const float sci = scl[0], scu = scl[1];
    acc_oi *= sci; acc_ou *= scu;
#pragma unroll
    for (int r = 0; r < 8; ++r) {
      int nm = r + (half << 3);
      acc_oi += esc[0][nm] * (aiv[r] + bvi);
      acc_ou += esc[1][nm] * (auv[r] + bvu);
    }
    __syncthreads();   // all reads of Xf[t&1] and esc complete

    // ---- TDM pipeline: refill the buffer just consumed, ensure next ready ----
    if (w == 0) {
      if (t + 2 < ntiles) {
        tdm_load_tile((t & 1) ? lds1 : lds0,
                      gbase + ((unsigned long long)(t + 2) * TM) * FDIM * 4,
                      (unsigned)(seg_len - (t + 2) * TM));
        __builtin_amdgcn_s_wait_tensorcnt(1);   // tile t+1 complete
      } else if (t + 1 < ntiles) {
        __builtin_amdgcn_s_wait_tensorcnt(0);   // tile t+1 complete
      }
    }
    __syncthreads();   // publish tile t+1 to all waves
  }

  // ---- finalize: combine lane halves, divide by expsum, store ----
  const float Si = srun[0], Su = srun[1];
  float oi = acc_oi + __shfl_xor(acc_oi, 16, 32);
  float ou = acc_ou + __shfl_xor(acc_ou, 16, 32);
  if (half == 0) {
    out[(size_t)b * HDIM + col]                     = oi / Si;
    out[(size_t)Bn * HDIM + (size_t)b * HDIM + col] = ou / Su;
  }
}

// ---------------------------------------------------------------------------
extern "C" void kernel_launch(void* const* d_in, const int* in_sizes, int n_in,
                              void* d_out, int out_size, void* d_ws, size_t ws_size,
                              hipStream_t stream) {
  (void)n_in; (void)ws_size;
  const float* feat = (const float*)d_in[0];
  const int*   seg  = (const int*)d_in[1];
  const float* i_Wq = (const float*)d_in[3];  const float* i_bq = (const float*)d_in[4];
  const float* i_Wk = (const float*)d_in[5];  const float* i_bk = (const float*)d_in[6];
  const float* i_Wv = (const float*)d_in[7];  const float* i_bv = (const float*)d_in[8];
  const float* u_Wq = (const float*)d_in[9];  const float* u_bq = (const float*)d_in[10];
  const float* u_Wk = (const float*)d_in[11]; const float* u_bk = (const float*)d_in[12];
  const float* u_Wv = (const float*)d_in[13]; const float* u_bv = (const float*)d_in[14];

  const int Bn = out_size / (2 * HDIM);   // 4096
  const int N  = in_sizes[1];             // 262144

  unsigned short* wpack = (unsigned short*)d_ws;                  // 6*16384 bf16
  int* first = (int*)((char*)d_ws + 6 * 16384 * sizeof(unsigned short));

  const float* mats[6] = { i_Wq, u_Wq, i_Wk, i_Wv, u_Wk, u_Wv };
  for (int m = 0; m < 6; ++m)
    pack_weight_kernel<<<64, 256, 0, stream>>>(mats[m], wpack + m * 16384);

  build_first_kernel<<<(Bn + 1 + 255) / 256, 256, 0, stream>>>(seg, first, Bn, N);

  seg_attn_kernel<<<Bn, 256, 0, stream>>>(feat, first, wpack,
                                          i_bq, i_bk, i_bv, u_bq, u_bk, u_bv,
                                          (float*)d_out, Bn);
}
